// Encoder_19232863551942
// MI455X (gfx1250) — compile-verified, matrix operand order checked by reading, not run on hardware
//
#include <hip/hip_runtime.h>
#include <hip/hip_bf16.h>
#include <cstdint>
#include <cstddef>

// ---- problem constants (match reference) ----
#define VOCABN 32000
#define EMBN   512
#define HIDN   512
#define BATCHN 32
#define SEQN   512
#define G3     1536          // 3*HID
#define NGATE  3072          // 2*3*HID

typedef __attribute__((ext_vector_type(16))) __bf16 v16bf;
typedef __attribute__((ext_vector_type(8)))  __bf16 v8bf;
typedef __attribute__((ext_vector_type(4)))  __bf16 v4bf;
typedef __attribute__((ext_vector_type(8)))  float  v8f;

// ---------------------------------------------------------------------------
// WMMA helpers.  CDNA5 16x16x32 bf16, f32 accumulate.
// A fragment layout (ISA 7.12.2, 16-bit A 16x32): lane L holds row m=L%16,
// K runs [kb,kb+8) and [16+kb,16+kb+8) with kb=(L<16)?0:8 -> two 16B LDS reads.
// B (KxN) stored K-major per N row uses the identical per-lane pattern.
// ---------------------------------------------------------------------------
__device__ inline v16bf load_frag_row(const __bf16* row_kb) {
    v8bf lo = *(const v8bf*)(row_kb);
    v8bf hi = *(const v8bf*)(row_kb + 16);
    return __builtin_shufflevector(lo, hi, 0,1,2,3,4,5,6,7,8,9,10,11,12,13,14,15);
}

__device__ inline v8f wmma_bf16(v16bf a, v16bf b, v8f c) {
    return __builtin_amdgcn_wmma_f32_16x16x32_bf16(
        false, a, false, b, (short)0, c, false, false);
}

// Async global->LDS DMA of one 16B chunk (CDNA5, tracked by ASYNCcnt).
__device__ inline void async_copy16(uint32_t lds_off, const void* gptr) {
    asm volatile("global_load_async_to_lds_b128 %0, %1, off"
                 :: "v"(lds_off), "v"((uint64_t)(uintptr_t)gptr) : "memory");
}
__device__ inline void wait_async0() {
    asm volatile("s_wait_asynccnt 0x0" ::: "memory");
}

// ---------------------------------------------------------------------------
// Kernel 1: f32 -> bf16 conversion of emb / W_ih / W_hh (vectorized float4 ->
// v_cvt_pk_bf16_f32) + zero the cross-WG sync counters.
// ---------------------------------------------------------------------------
#define EMB4 (VOCABN * EMBN / 4)     // 4,096,000 float4 groups
#define W4   (2 * G3 * EMBN / 4)     // 393,216 float4 groups

__global__ __launch_bounds__(256) void cvt_kernel(
    const float4* __restrict__ emb4, const float4* __restrict__ Wih4,
    const float4* __restrict__ Whh4, v4bf* __restrict__ embb4,
    v4bf* __restrict__ Wihb4, v4bf* __restrict__ Whhb4,
    uint32_t* __restrict__ cnt)
{
    const int idx = blockIdx.x * 256 + threadIdx.x;
    if (idx < 2) cnt[idx] = 0u;
    if (idx < EMB4) {
        const float4 f = emb4[idx];
        v4bf r = { (__bf16)f.x, (__bf16)f.y, (__bf16)f.z, (__bf16)f.w };
        embb4[idx] = r;
    }
    if (idx < W4) {
        const float4 a = Wih4[idx];
        v4bf ra = { (__bf16)a.x, (__bf16)a.y, (__bf16)a.z, (__bf16)a.w };
        Wihb4[idx] = ra;
        const float4 b = Whh4[idx];
        v4bf rb = { (__bf16)b.x, (__bf16)b.y, (__bf16)b.z, (__bf16)b.w };
        Whhb4[idx] = rb;
    }
}

// ---------------------------------------------------------------------------
// Kernel 2: gx = gather(emb_bf16, enc) @ W_ih^T + b_ih  (M=16384,N=3072,K=512)
// Block tile 128x128x32, 256 threads (8 waves): wave w owns N-tile w x all 8
// M-tiles (8 v8f accumulators).  Tiles staged by double-buffered async
// global->LDS DMA; all 9 fragments preloaded into distinct registers before
// the 8-WMMA chain so ds_loads overlap the matrix pipe.
// ---------------------------------------------------------------------------
__global__ __launch_bounds__(256) void gx_gemm_kernel(
    const int* __restrict__ enc, const __bf16* __restrict__ embb,
    const __bf16* __restrict__ Wih, const float* __restrict__ bih,
    __bf16* __restrict__ gx)
{
    __shared__ __bf16 Als[2][128 * 40];   // 128 rows x 32k (+8 pad) x 2 bufs
    __shared__ __bf16 Bls[2][128 * 40];
    __shared__ int    tok[128];

    const int tid  = threadIdx.x;
    const int lane = tid & 31;
    const int wave = tid >> 5;          // 0..7 -> N-tile
    const int bm   = blockIdx.x;        // 0..127
    const int bn   = blockIdx.y;        // 0..23

    if (tid < 128) tok[tid] = enc[bm * 128 + tid];
    __syncthreads();

    // 1024 16B chunks per k-step: 512 for A (gathered emb rows), 512 for B.
    auto issue_tile = [&](int p, int k0) {
        for (int c = tid; c < 1024; c += 256) {
            if (c < 512) {
                const int row = c >> 2, q = c & 3;          // q*8 bf16 within row
                async_copy16((uint32_t)(uintptr_t)&Als[p][row * 40 + q * 8],
                             embb + (size_t)tok[row] * EMBN + k0 + q * 8);
            } else {
                const int row = (c - 512) >> 2, q = c & 3;
                async_copy16((uint32_t)(uintptr_t)&Bls[p][row * 40 + q * 8],
                             Wih + (size_t)(bn * 128 + row) * EMBN + k0 + q * 8);
            }
        }
    };

    v8f acc[8] = {};
    const int kb = (lane < 16) ? 0 : 8;

    issue_tile(0, 0);
    for (int kt = 0; kt < 16; ++kt) {
        const int p = kt & 1;
        wait_async0();          // own-wave DMA for tile kt complete
        __syncthreads();        // all waves' DMA done; prior reads of buf p done
        if (kt < 15) issue_tile(p ^ 1, (kt + 1) * 32);   // overlap next DMA

        // preload all fragments into distinct registers, then the WMMA chain
        const v16bf bf = load_frag_row(&Bls[p][(wave * 16 + (lane & 15)) * 40 + kb]);
        v16bf af[8];
        #pragma unroll
        for (int i = 0; i < 8; ++i)
            af[i] = load_frag_row(&Als[p][(i * 16 + (lane & 15)) * 40 + kb]);
        #pragma unroll
        for (int i = 0; i < 8; ++i)
            acc[i] = wmma_bf16(af[i], bf, acc[i]);
    }

    // Epilogue: C layout = lanes 0-15: M=v, N=lane; lanes 16-31: M=8+v, N=lane-16
    const int ncol   = bn * 128 + wave * 16 + (lane & 15);
    const float bias = bih[ncol];
    const int moff   = (lane < 16) ? 0 : 8;
    #pragma unroll
    for (int i = 0; i < 8; ++i) {
        #pragma unroll
        for (int v = 0; v < 8; ++v) {
            const int grow = bm * 128 + i * 16 + moff + v;
            gx[(size_t)grow * NGATE + ncol] = (__bf16)(acc[i][v] + bias);
        }
    }
}

// ---------------------------------------------------------------------------
// Kernel 3: persistent bidirectional GRU scan.
// Grid = 32 WGs: dir = blk/16, slice = blk%16 (32 h-columns -> 96 gate rows).
// W_hh slice (96x512 bf16, ~100 KB) resident in LDS for all 512 steps, loaded
// once via async DMA.  Per step: 6 waves x 2 M-tiles of wmma bf16 with a
// one-stage software pipeline on the k-loop, f32 gates, then a device-scope
// barrier across the direction's 16 WGs.
// ---------------------------------------------------------------------------
#define WROW 520                      // padded LDS row stride (bf16 elems)
#define SMEM_SCAN (96*WROW*2 + 32*WROW*2 + 32*100*4)

__global__ __launch_bounds__(192) void gru_scan_kernel(
    const __bf16* __restrict__ gx, const __bf16* __restrict__ Whh,
    const float* __restrict__ bhh, float* __restrict__ out,
    float* __restrict__ hbuf, uint32_t* __restrict__ cnt)
{
    extern __shared__ char smem[];
    __bf16* Wsl = (__bf16*)smem;                                   // 96 x WROW
    __bf16* Hls = (__bf16*)(smem + 96 * WROW * 2);                 // 32 x WROW
    float*  Ghl = (float*) (smem + 96 * WROW * 2 + 32 * WROW * 2); // 32 x 100

    const int tid   = threadIdx.x;
    const int lane  = tid & 31;
    const int wave  = tid >> 5;        // 0..5 -> N-tile of the 96 gate cols
    const int dir   = blockIdx.x >> 4;
    const int slice = blockIdx.x & 15;

    // one-time: async DMA the W_hh slice (rows g*512 + slice*32 + c) into LDS
    const __bf16* Wd = Whh + (size_t)dir * G3 * HIDN;
    for (int c = tid; c < 96 * 64; c += 192) {       // 64 x 16B chunks per row
        const int row = c >> 6, off = c & 63;
        const int grow = (row >> 5) * HIDN + slice * 32 + (row & 31);
        async_copy16((uint32_t)(uintptr_t)(&Wsl[row * WROW]) + off * 16,
                     (const char*)(Wd + (size_t)grow * HIDN) + off * 16);
    }
    wait_async0();
    __syncthreads();

    float* hb = hbuf + (size_t)dir * 2 * BATCHN * HIDN;
    const int kb = (lane < 16) ? 0 : 8;
    const int m0 = (lane & 15) * WROW;
    const int m1 = (16 + (lane & 15)) * WROW;
    const int nr = (wave * 16 + (lane & 15)) * WROW;

    for (int t = 0; t < SEQN; ++t) {
        // stage h_{t-1} into LDS as bf16 (zeros at t=0)
        if (t == 0) {
            for (int q = tid; q < 32 * 128; q += 192) {
                const int b = q >> 7, k4 = q & 127;
                v4bf z = {};
                *(v4bf*)&Hls[b * WROW + k4 * 4] = z;
            }
        } else {
            const float* hsrc = hb + ((t - 1) & 1) * BATCHN * HIDN;
            for (int q = tid; q < 32 * 128; q += 192) {
                const int b = q >> 7, k4 = q & 127;
                const float4 f = *(const float4*)&hsrc[b * HIDN + k4 * 4];
                v4bf z = { (__bf16)f.x, (__bf16)f.y, (__bf16)f.z, (__bf16)f.w };
                *(v4bf*)&Hls[b * WROW + k4 * 4] = z;
            }
        }
        __syncthreads();

        // gh = h_{t-1} @ Whh_slice^T : M=32, N=96 (wave owns 16), K=512
        // one-stage pipeline: fragments for kt+1 load under the kt WMMAs
        v8f acc0 = {}, acc1 = {};
        v16bf bf  = load_frag_row(&Wsl[nr + kb]);
        v16bf af0 = load_frag_row(&Hls[m0 + kb]);
        v16bf af1 = load_frag_row(&Hls[m1 + kb]);
        #pragma unroll 4
        for (int kt = 0; kt < 16; ++kt) {
            v16bf nbf, naf0, naf1;
            if (kt < 15) {
                const int k0 = (kt + 1) * 32;
                nbf  = load_frag_row(&Wsl[nr + k0 + kb]);
                naf0 = load_frag_row(&Hls[m0 + k0 + kb]);
                naf1 = load_frag_row(&Hls[m1 + k0 + kb]);
            }
            acc0 = wmma_bf16(af0, bf, acc0);
            acc1 = wmma_bf16(af1, bf, acc1);
            bf = nbf; af0 = naf0; af1 = naf1;
        }
        const int gcol = wave * 16 + (lane & 15);
        const int roff = (lane < 16) ? 0 : 8;
        #pragma unroll
        for (int v = 0; v < 8; ++v) {
            Ghl[(roff + v) * 100 + gcol]      = acc0[v];
            Ghl[(16 + roff + v) * 100 + gcol] = acc1[v];
        }
        __syncthreads();

        // gates + outputs
        const float* hsrc = hb + ((t - 1) & 1) * BATCHN * HIDN;
        float*       hdst = hb + (t & 1) * BATCHN * HIDN;
        for (int e = tid; e < 1024; e += 192) {
            const int b = e >> 5, c = e & 31;
            const int col = slice * 32 + c;
            const float ghr = Ghl[b * 100 + c]      + bhh[dir * G3 + col];
            const float ghz = Ghl[b * 100 + 32 + c] + bhh[dir * G3 + 512 + col];
            const float ghn = Ghl[b * 100 + 64 + c] + bhh[dir * G3 + 1024 + col];
            const size_t gxb = (size_t)(b * SEQN + t) * NGATE + dir * G3;
            const float gxr = (float)gx[gxb + col];
            const float gxz = (float)gx[gxb + 512 + col];
            const float gxn = (float)gx[gxb + 1024 + col];
            const float hp  = (t == 0) ? 0.f : hsrc[b * HIDN + col];
            const float r  = 1.f / (1.f + __expf(-(gxr + ghr)));
            const float z  = 1.f / (1.f + __expf(-(gxz + ghz)));
            const float n  = tanhf(gxn + r * ghn);
            const float hn = (1.f - z) * n + z * hp;
            out[(size_t)(b * SEQN + t) * 1024 + dir * HIDN + col] = hn;
            hdst[b * HIDN + col] = hn;
            if (t == SEQN - 1)
                out[(size_t)BATCHN * SEQN * 1024 + (dir * BATCHN + b) * HIDN + col] = hn;
        }
        __syncthreads();

        // device-scope barrier across the 16 WGs of this direction
        __threadfence();
        if (tid == 0) {
            atomicAdd(&cnt[dir], 1u);
            const uint32_t target = 16u * (uint32_t)(t + 1);
            while (__hip_atomic_load(&cnt[dir], __ATOMIC_ACQUIRE,
                                     __HIP_MEMORY_SCOPE_AGENT) < target) { }
        }
        __syncthreads();
    }
}

// ---------------------------------------------------------------------------
// Host launcher
// ---------------------------------------------------------------------------
extern "C" void kernel_launch(void* const* d_in, const int* in_sizes, int n_in,
                              void* d_out, int out_size, void* d_ws, size_t ws_size,
                              hipStream_t stream) {
    const int*   enc = (const int*)d_in[0];
    const float* emb = (const float*)d_in[1];
    const float* Wih = (const float*)d_in[2];
    const float* Whh = (const float*)d_in[3];
    const float* bih = (const float*)d_in[4];
    const float* bhh = (const float*)d_in[5];
    float* out = (float*)d_out;

    char* ws = (char*)d_ws;
    // layout: [cnt 256B][Wih bf16 3MB][Whh bf16 3MB][emb bf16 32.8MB]
    //         [hbuf f32 256KB][gx bf16 ~100MB]   (total ~134 MB)
    uint32_t* cnt   = (uint32_t*)ws;
    __bf16*   Wih_b = (__bf16*)(ws + 256);
    __bf16*   Whh_b = (__bf16*)(ws + 256 + 3145728);
    __bf16*   emb_b = (__bf16*)(ws + 256 + 2 * 3145728);
    float*    hbuf  = (float*) (ws + 256 + 2 * 3145728 + 32768000);
    __bf16*   gx    = (__bf16*)(ws + 256 + 2 * 3145728 + 32768000 + 262144);

    cvt_kernel<<<EMB4 / 256, 256, 0, stream>>>(
        (const float4*)emb, (const float4*)Wih, (const float4*)Whh,
        (v4bf*)emb_b, (v4bf*)Wih_b, (v4bf*)Whh_b, cnt);

    gx_gemm_kernel<<<dim3(128, 24), 256, 0, stream>>>(enc, emb_b, Wih_b, bih, gx);

    (void)hipFuncSetAttribute((const void*)gru_scan_kernel,
                              hipFuncAttributeMaxDynamicSharedMemorySize,
                              SMEM_SCAN);
    gru_scan_kernel<<<32, 192, SMEM_SCAN, stream>>>(gx, Whh_b, bhh, out, hbuf, cnt);
}
